// CompressedLinearFP32_67920612819207
// MI455X (gfx1250) — matrix operand order repaired and checked by reading, not compiled, not run on hardware
//
#include <hip/hip_runtime.h>

// CDNA5 / gfx1250 WMMA int8-weight linear:
//   out[M,N] = f16(x) @ (f16(w_int8) * f16(scale))^T + bias, fp32 accum
// M=8192 (B*S), K=4096 (IN), N=11008 (OUT)
//
// Block tile 256x128, BK=32, 8 wave32s in 4(M) x 2(N) grid, wave tile 64x64:
// 16 v_wmma_f32_16x16x32_f16 per wave per K-step, double-buffered LDS.

typedef __attribute__((ext_vector_type(16))) _Float16 v16h;
typedef __attribute__((ext_vector_type(8)))  _Float16 v8h;
typedef __attribute__((ext_vector_type(8)))  float    v8f;
typedef __attribute__((ext_vector_type(4)))  float    v4f;
typedef __attribute__((ext_vector_type(4)))  int      v4i;

#define M_DIM 8192
#define N_DIM 11008
#define K_DIM 4096

#define BM 256
#define BN 128
#define BK 32
#define NKT (K_DIM / BK)   // 128 K-tiles

__global__ __launch_bounds__(256)
void qlinear_wmma_f16_kernel(const float* __restrict__ x,
                             const int*   __restrict__ w,      // int8 values stored as int32
                             const float* __restrict__ scale,  // per-OUT-row scale
                             const float* __restrict__ bias,
                             float*       __restrict__ out)
{
    // Double-buffered LDS tiles (f16, K contiguous per row)
    __shared__ _Float16 As[2][BM * BK];   // 16 KB each
    __shared__ _Float16 Bs[2][BN * BK];   // 8 KB each

    const int tid  = threadIdx.x;
    const int lane = tid & 31;
    const int wave = tid >> 5;       // 0..7
    const int wm   = wave & 3;       // 0..3 -> 64-row slab
    const int wn   = wave >> 2;      // 0..1 -> 64-col slab

    const int bm = blockIdx.y * BM;
    const int bn = blockIdx.x * BN;

    // ---- loader coordinates ----
    // A: one x-row per thread, full BK=32 floats (8 x b128)
    const int arow = tid;                       // 0..255
    // B: half a weight-row per thread, 16 ints (4 x b128)
    const int brow = tid >> 1;                  // 0..127
    const int bk   = (tid & 1) * 16;            // 0 or 16

    const v4f* __restrict__ xp = (const v4f*)(x + (size_t)(bm + arow) * K_DIM);
    const v4i* __restrict__ wp = (const v4i*)(w + (size_t)(bn + brow) * K_DIM + bk);
    const _Float16 sh = (_Float16)scale[bn + brow];   // per-row scale in f16 (matches ref)

    // ---- accumulators: 4 (M) x 4 (N) 16x16 f32 tiles per wave (128 VGPRs) ----
    v8f acc[4][4];
#pragma unroll
    for (int i = 0; i < 4; ++i)
#pragma unroll
        for (int j = 0; j < 4; ++j)
            acc[i][j] = (v8f){0.f, 0.f, 0.f, 0.f, 0.f, 0.f, 0.f, 0.f};

    const int r  = lane & 15;          // M row (A/C) or N col (B/C) within a fragment
    const int h8 = (lane >> 4) * 8;    // lane-half K offset

    // staging registers for global->LDS with in-register conversion
    v4f ax[8];
    v4i bw[4];

    // ---------- prologue: fetch K-tile 0 and stage into buffer 0 ----------
#pragma unroll
    for (int i = 0; i < 8; ++i) ax[i] = xp[i];
#pragma unroll
    for (int i = 0; i < 4; ++i) bw[i] = wp[i];

    {
        _Float16 ah[32], bh[16];
#pragma unroll
        for (int i = 0; i < 8; ++i)
#pragma unroll
            for (int j = 0; j < 4; ++j)
                ah[i * 4 + j] = (_Float16)ax[i][j];
#pragma unroll
        for (int i = 0; i < 4; ++i)
#pragma unroll
            for (int j = 0; j < 4; ++j)
                bh[i * 4 + j] = (_Float16)bw[i][j] * sh;   // dequant in f16, like reference

        _Float16* ad = &As[0][arow * BK];
#pragma unroll
        for (int c = 0; c < 4; ++c)
            *(v8h*)(ad + c * 8) = *(v8h*)&ah[c * 8];
        _Float16* bd = &Bs[0][brow * BK + bk];
        *(v8h*)(bd)     = *(v8h*)&bh[0];
        *(v8h*)(bd + 8) = *(v8h*)&bh[8];
    }
    __syncthreads();

    // ---------- main K loop, double buffered ----------
    for (int kt = 0; kt < NKT; ++kt) {
        const int  buf  = kt & 1;
        const bool more = (kt + 1) < NKT;

        // issue global loads for the NEXT K-tile first (overlap with WMMA below)
        if (more) {
            const size_t off = (size_t)(kt + 1) * 8;   // 8 x v4f per BK step
#pragma unroll
            for (int i = 0; i < 8; ++i) ax[i] = xp[off + i];
#pragma unroll
            for (int i = 0; i < 4; ++i) bw[i] = wp[off + i];
        }

        // ---- fragment loads from LDS (contiguous K order, same for A and B) ----
        // A 16x32 f16: lanes 0-15 hold K 0..15, lanes 16-31 hold K 16..31
        v16h afrag[4];
#pragma unroll
        for (int sm = 0; sm < 4; ++sm) {
            const _Float16* p = &As[buf][(wm * 64 + sm * 16 + r) * BK];
            v8h lo = *(const v8h*)(p + h8 * 2);
            v8h hi = *(const v8h*)(p + 8 + h8 * 2);
            v16h a;
#pragma unroll
            for (int e = 0; e < 8; ++e) { a[e] = lo[e]; a[8 + e] = hi[e]; }
            afrag[sm] = a;
        }
        // B 32x16 f16: lane = N col; lanes 0-15 hold K 0..15, lanes 16-31 hold K 16..31
        v16h bfrag[4];
#pragma unroll
        for (int sn = 0; sn < 4; ++sn) {
            const _Float16* p = &Bs[buf][(wn * 64 + sn * 16 + r) * BK + h8 * 2];
            bfrag[sn] = *(const v16h*)p;
        }

        // ---- 16 back-to-back WMMAs: D = A x B + C ----
#pragma unroll
        for (int sm = 0; sm < 4; ++sm)
#pragma unroll
            for (int sn = 0; sn < 4; ++sn)
                acc[sm][sn] = __builtin_amdgcn_wmma_f32_16x16x32_f16(
                    /*neg_a=*/false, afrag[sm],
                    /*neg_b=*/false, bfrag[sn],
                    /*c_mod=*/(short)0, acc[sm][sn],
                    /*reuse_a=*/false, /*reuse_b=*/false);

        // ---- convert and stage next tile into the other buffer ----
        if (more) {
            _Float16 ah[32], bh[16];
#pragma unroll
            for (int i = 0; i < 8; ++i)
#pragma unroll
                for (int j = 0; j < 4; ++j)
                    ah[i * 4 + j] = (_Float16)ax[i][j];
#pragma unroll
            for (int i = 0; i < 4; ++i)
#pragma unroll
                for (int j = 0; j < 4; ++j)
                    bh[i * 4 + j] = (_Float16)bw[i][j] * sh;

            _Float16* ad = &As[buf ^ 1][arow * BK];
#pragma unroll
            for (int c = 0; c < 4; ++c)
                *(v8h*)(ad + c * 8) = *(v8h*)&ah[c * 8];
            _Float16* bd = &Bs[buf ^ 1][brow * BK + bk];
            *(v8h*)(bd)     = *(v8h*)&bh[0];
            *(v8h*)(bd + 8) = *(v8h*)&bh[8];
        }
        __syncthreads();
    }

    // ---------- epilogue: bias add + store ----------
    // C/D layout: VGPR v, lane l -> M = v + 8*(l>=16), N = l%16
#pragma unroll
    for (int sn = 0; sn < 4; ++sn) {
        const int gn = bn + wn * 64 + sn * 16 + r;
        const float bv = bias[gn];
#pragma unroll
        for (int sm = 0; sm < 4; ++sm) {
            const int mbase = bm + wm * 64 + sm * 16 + (lane >> 4) * 8;
            v8f c = acc[sm][sn];
#pragma unroll
            for (int v = 0; v < 8; ++v) {
                out[(size_t)(mbase + v) * N_DIM + gn] = c[v] + bv;
            }
        }
    }
}

extern "C" void kernel_launch(void* const* d_in, const int* in_sizes, int n_in,
                              void* d_out, int out_size, void* d_ws, size_t ws_size,
                              hipStream_t stream) {
    (void)in_sizes; (void)n_in; (void)out_size; (void)d_ws; (void)ws_size;
    const float* x     = (const float*)d_in[0];
    const int*   w     = (const int*)  d_in[1];
    const float* scale = (const float*)d_in[2];
    const float* bias  = (const float*)d_in[3];
    float*       out   = (float*)d_out;

    dim3 grid(N_DIM / BN, M_DIM / BM);   // (86, 32)
    dim3 block(256);                     // 8 wave32s
    qlinear_wmma_f16_kernel<<<grid, block, 0, stream>>>(x, w, scale, bias, out);
}
